// Mamba2Forecast_74646531604910
// MI455X (gfx1250) — compile-verified, matrix operand order checked by reading, not run on hardware
//
#include <hip/hip_runtime.h>

// Mamba2 forecast, MI455X (gfx1250, wave32). Key structural wins:
//  (1) only last-token output is consumed -> skip z0/x0/z for all but t=L-1
//  (2) dA is per-head scalar -> scan collapses to log-space suffix-sum + weighted sum
//  (3) the single real GEMM (16384x128 @ 128x386) runs on v_wmma_f32_16x16x32_f16,
//      each wave computing a 16x64 strip (A loaded once per K-step, 16 WMMAs/wave)

typedef __attribute__((ext_vector_type(16))) _Float16 v16h;
typedef __attribute__((ext_vector_type(8)))  _Float16 v8h;
typedef __attribute__((ext_vector_type(8)))  float    v8f;

#define BATCH     8
#define SEQLEN    2048
#define IN_SIZE   8
#define D_MODEL   128
#define D_IN_PROJ 770
#define NTOK      (BATCH * SEQLEN)        // 16384
#define XCOLS     386                     // xBC(384) + dt(2) = in_proj cols [384,770)
#define XSTRIDE   388                     // padded row stride (16B-aligned rows)
#define NTILE_PAD 28                      // 25 live N-tiles, padded to 7 groups of 4
#define NGROUPS   7
#define KTILES    4                       // K=128 / 32

// workspace layout (bytes); total ~29.7 MB
#define OFF_H16   0u
#define SZ_H16    (NTOK * D_MODEL * 2)                    // 4,194,304
#define OFF_BPACK (OFF_H16 + SZ_H16)
#define SZ_BPACK  (KTILES * NTILE_PAD * 32 * 16 * 2)      // 114,688
#define OFF_XBC   (OFF_BPACK + SZ_BPACK)                  // 16B aligned
#define SZ_XBC    (NTOK * XSTRIDE * 4)                    // 25,427,968
#define OFF_Y     (OFF_XBC + SZ_XBC)

__device__ __forceinline__ float siluf(float v) { return v / (1.f + __expf(-v)); }

// ---- kernel 1: h = x @ emb_W + emb_b, cast to f16 -------------------------
__global__ void k_embed(const float* __restrict__ x, const float* __restrict__ embW,
                        const float* __restrict__ embB, _Float16* __restrict__ h16) {
  int idx = blockIdx.x * blockDim.x + threadIdx.x;   // NTOK*128 threads
  int tok = idx >> 7, d = idx & 127;
  const float* xr = x + tok * IN_SIZE;
  float acc = embB[d];
#pragma unroll
  for (int k = 0; k < IN_SIZE; ++k) acc += xr[k] * embW[k * D_MODEL + d];
  h16[idx] = (_Float16)acc;
}

// ---- kernel 2: pack in_proj_W cols [384,770) into WMMA B-operand layout ----
// per-lane layout for 16-bit 32x16 B tile: n = lane&15,
// half i -> K = (lane>>4)*8 + (i<8 ? i : i+8)   (mirrors the 16-bit A layout)
// N-tiles 25..27 are zero padding so the GEMM inner loop needs no guards.
__global__ void k_pack(const float* __restrict__ W, _Float16* __restrict__ bp) {
  int nt = blockIdx.x;            // 0..27
  int kt = blockIdx.y;            // 0..3
  int l  = threadIdx.x;           // 0..31
  int n  = 384 + nt * 16 + (l & 15);
  int kb = kt * 32 + (l >> 4) * 8;
  _Float16* dst = bp + (((kt * NTILE_PAD) + nt) * 32 + l) * 16;
#pragma unroll
  for (int i = 0; i < 16; ++i) {
    int K = kb + (i < 8 ? i : i + 8);
    float v = (n < D_IN_PROJ) ? W[K * D_IN_PROJ + n] : 0.f;
    dst[i] = (_Float16)v;
  }
}

// ---- kernel 3: WMMA GEMM: xbc[16384 x 386] = h16[16384x128] @ W16[128x386] -
// Each wave: one 16(M) x 64(N) strip; A loaded once per K-step, 16 WMMAs total.
__global__ void k_gemm_wmma(const _Float16* __restrict__ h16,
                            const _Float16* __restrict__ bp,
                            float* __restrict__ xbc) {
  int wave  = threadIdx.x >> 5;
  int l     = threadIdx.x & 31;
  int mtile = blockIdx.x * 8 + wave;       // 0..1023
  int ng    = blockIdx.y;                  // 0..6 (N-tile group of 4)
  int m  = l & 15;
  int hi = l >> 4;
  int row0 = mtile * 16;
  // 16-bit A 16x32 layout: lanes 0-15 hold K{0-7,16-23}, lanes 16-31 K{8-15,24-31}
  const _Float16* arow  = h16 + (row0 + m) * D_MODEL + hi * 8;
  const _Float16* bbase = bp + ((ng * 4) * 32 + l) * 16;
  v8f c0 = {}, c1 = {}, c2 = {}, c3 = {};
#pragma unroll
  for (int kt = 0; kt < KTILES; ++kt) {
    const _Float16* ap = arow + kt * 32;
    if (kt < KTILES - 1) __builtin_prefetch(ap + 32, 0, 1);  // global_prefetch_b8
    union { v16h v; v8h h[2]; } a;
    a.h[0] = *(const v8h*)(ap);            // K 0..7   (+kt*32, +hi*8)
    a.h[1] = *(const v8h*)(ap + 16);       // K 16..23
    const _Float16* bk = bbase + kt * (NTILE_PAD * 32 * 16);
    union { v16h v; v8h h[2]; } b0, b1, b2, b3;
    b0.h[0] = *(const v8h*)(bk);           b0.h[1] = *(const v8h*)(bk + 8);
    b1.h[0] = *(const v8h*)(bk + 512);     b1.h[1] = *(const v8h*)(bk + 520);
    b2.h[0] = *(const v8h*)(bk + 1024);    b2.h[1] = *(const v8h*)(bk + 1032);
    b3.h[0] = *(const v8h*)(bk + 1536);    b3.h[1] = *(const v8h*)(bk + 1544);
    c0 = __builtin_amdgcn_wmma_f32_16x16x32_f16(false, a.v, false, b0.v, (short)0, c0, false, false);
    c1 = __builtin_amdgcn_wmma_f32_16x16x32_f16(false, a.v, false, b1.v, (short)0, c1, false, false);
    c2 = __builtin_amdgcn_wmma_f32_16x16x32_f16(false, a.v, false, b2.v, (short)0, c2, false, false);
    c3 = __builtin_amdgcn_wmma_f32_16x16x32_f16(false, a.v, false, b3.v, (short)0, c3, false, false);
  }
  // D layout: N = lane&15, M = v + 8*(lane>>4); guard the 386-col ragged edge
  v8f acc[4] = {c0, c1, c2, c3};
#pragma unroll
  for (int j = 0; j < 4; ++j) {
    int col = (ng * 4 + j) * 16 + m;
    if (col < XCOLS) {
#pragma unroll
      for (int v = 0; v < 8; ++v)
        xbc[(row0 + v + hi * 8) * XSTRIDE + col] = acc[j][v];
    }
  }
}

// ---- kernel 4: fused conv + collapsed scan, one block per (batch, head) ----
__global__ void k_scan(const float* __restrict__ xbc,
                       const float* __restrict__ convW, const float* __restrict__ convB,
                       const float* __restrict__ dt_bias, const float* __restrict__ A_log,
                       const float* __restrict__ Dp, float* __restrict__ yhead) {
  int bh = blockIdx.x, b = bh >> 1, h = bh & 1;
  int tid = threadIdx.x;                   // 256
  const int tokbase = b * SEQLEN;

  __shared__ float Clast[128];
  __shared__ float sd[SEQLEN];             // dt_t * (B_t . C_last)
  __shared__ float ga[SEQLEN];             // log a_t, then g_t
  __shared__ float tot[256];
  __shared__ float part[256];

  const float expA = __expf(A_log[h]);
  const float dtb  = dt_bias[h];

  auto conv = [&](int t, int c) -> float {  // depthwise conv (4 taps) + silu
    float s = convB[c];
    const float* cw = convW + c * 4;
#pragma unroll
    for (int k = 0; k < 4; ++k) {
      int tt = t - 3 + k;
      if (tt >= 0) s += xbc[(tokbase + tt) * XSTRIDE + c] * cw[k];
    }
    return siluf(s);
  };

  if (tid < 128) Clast[tid] = conv(SEQLEN - 1, 256 + tid);
  __syncthreads();

  // pass 1: per-t log(a_t) and dt_t*(B_t.C_last); reversed index r = L-1-t
  float csum = 0.f;
  for (int i = 0; i < 8; ++i) {
    int r = tid * 8 + i, t = SEQLEN - 1 - r;
    float v   = xbc[(tokbase + t) * XSTRIDE + 384 + h] + dtb;
    float dts = (v > 20.f) ? v : __logf(1.f + __expf(v));   // softplus
    float s = 0.f;
    for (int n = 0; n < 128; ++n) s += conv(t, 128 + n) * Clast[n];
    sd[t] = dts * s;
    ga[t] = -dts * expA;
    csum += -dts * expA;
  }
  tot[tid] = csum;
  __syncthreads();
  if (tid == 0) {                           // exclusive scan over chunk totals
    float run = 0.f;
    for (int j = 0; j < 256; ++j) { float tv = tot[j]; tot[j] = run; run += tv; }
  }
  __syncthreads();
  // pass 2: g_t = exp(suffix sum of log a, exclusive) * sd[t]
  float run = tot[tid];
  for (int i = 0; i < 8; ++i) {
    int r = tid * 8 + i, t = SEQLEN - 1 - r;
    float la = ga[t];
    ga[t] = __expf(run) * sd[t];
    run += la;
  }
  __syncthreads();
  // stage B: y[p] = sum_t g_t * x_t[p] + D*x_{L-1}[p], p-parallel, 4-way t-split
  int p = tid & 63, q = tid >> 6;
  int ch = h * 64 + p;
  float acc = 0.f;
  for (int t = q * 512; t < q * 512 + 512; ++t) acc += ga[t] * conv(t, ch);
  if (q == 0) acc += Dp[h] * conv(SEQLEN - 1, ch);
  part[tid] = acc;
  __syncthreads();
  if (q == 0)
    yhead[bh * 64 + p] = part[p] + part[64 + p] + part[128 + p] + part[192 + p];
}

// ---- kernel 5: last-token epilogue, one block (128 thr) per batch ---------
__global__ void k_head(const float* __restrict__ x, const float* __restrict__ embW,
                       const float* __restrict__ embB, const float* __restrict__ inW,
                       const float* __restrict__ normw, const float* __restrict__ yhead,
                       const float* __restrict__ outW, const float* __restrict__ fc1W,
                       const float* __restrict__ fc1b, const float* __restrict__ fc2W,
                       const float* __restrict__ fc2b, float* __restrict__ out) {
  int b = blockIdx.x, tid = threadIdx.x;   // 128
  __shared__ float hl[128], u[256], red[128], o1[128], o2[128];

  {  // emb of last token in f32
    const float* xr = x + (b * SEQLEN + SEQLEN - 1) * IN_SIZE;
    float acc = embB[tid];
#pragma unroll
    for (int k = 0; k < IN_SIZE; ++k) acc += xr[k] * embW[k * D_MODEL + tid];
    hl[tid] = acc;
  }
  __syncthreads();
  float z0 = 0.f, x0 = 0.f, zz = 0.f;      // in_proj cols [0,384) for t=L-1 only
  for (int k = 0; k < 128; ++k) {
    float hv = hl[k];
    const float* wr = inW + k * D_IN_PROJ;
    z0 += hv * wr[tid];
    x0 += hv * wr[128 + tid];
    zz += hv * wr[256 + tid];
  }
  float yh = yhead[(b * 2 + (tid >> 6)) * 64 + (tid & 63)];
  float yg = yh * siluf(zz);
  red[tid] = yg * yg;
  __syncthreads();
  for (int off = 64; off > 0; off >>= 1) { if (tid < off) red[tid] += red[tid + off]; __syncthreads(); }
  float yn = yg * rsqrtf(red[0] / 128.f + 1e-5f) * normw[tid];
  __syncthreads();
  u[tid]       = siluf(z0) * x0;
  u[128 + tid] = yn;
  __syncthreads();
  float a1 = 0.f;
  for (int c = 0; c < 256; ++c) a1 += u[c] * outW[c * D_MODEL + tid];
  o1[tid] = a1;
  __syncthreads();
  float a2 = fc1b[tid];
  for (int c = 0; c < 128; ++c) a2 += o1[c] * fc1W[c * D_MODEL + tid];
  o2[tid] = fmaxf(a2, 0.f);
  __syncthreads();
  red[tid] = o2[tid] * fc2W[tid];
  __syncthreads();
  for (int off = 64; off > 0; off >>= 1) { if (tid < off) red[tid] += red[tid + off]; __syncthreads(); }
  if (tid == 0) out[b] = red[0] + fc2b[0];
}

extern "C" void kernel_launch(void* const* d_in, const int* in_sizes, int n_in,
                              void* d_out, int out_size, void* d_ws, size_t ws_size,
                              hipStream_t stream) {
  const float* x     = (const float*)d_in[0];
  const float* embW  = (const float*)d_in[1];
  const float* embB  = (const float*)d_in[2];
  const float* inW   = (const float*)d_in[3];
  const float* convW = (const float*)d_in[4];
  const float* convB = (const float*)d_in[5];
  const float* dtb   = (const float*)d_in[6];
  const float* Alog  = (const float*)d_in[7];
  const float* Dp    = (const float*)d_in[8];
  const float* normw = (const float*)d_in[9];
  const float* outW  = (const float*)d_in[10];
  const float* fc1W  = (const float*)d_in[11];
  const float* fc1b  = (const float*)d_in[12];
  const float* fc2W  = (const float*)d_in[13];
  const float* fc2b  = (const float*)d_in[14];
  float* out = (float*)d_out;

  char* ws = (char*)d_ws;                  // needs ~29.7 MB
  _Float16* h16   = (_Float16*)(ws + OFF_H16);
  _Float16* bpack = (_Float16*)(ws + OFF_BPACK);
  float*    xbc   = (float*)(ws + OFF_XBC);
  float*    yhead = (float*)(ws + OFF_Y);

  k_embed<<<NTOK * D_MODEL / 256, 256, 0, stream>>>(x, embW, embB, h16);
  k_pack<<<dim3(NTILE_PAD, KTILES), 32, 0, stream>>>(inW, bpack);
  k_gemm_wmma<<<dim3(NTOK / 16 / 8, NGROUPS), 256, 0, stream>>>(h16, bpack, xbc);
  k_scan<<<BATCH * 2, 256, 0, stream>>>(xbc, convW, convB, dtb, Alog, Dp, yhead);
  k_head<<<BATCH, 128, 0, stream>>>(x, embW, embB, inW, normw, yhead,
                                    outW, fc1W, fc1b, fc2W, fc2b, out);
}
